// Backbone_45320494907484
// MI455X (gfx1250) — matrix operand order repaired and checked by reading, not compiled, not run on hardware
//
#include <hip/hip_runtime.h>
#include <hip/hip_bf16.h>
#include <cstdint>

// ---------------------------------------------------------------------------
// Problem constants (from reference)
// ---------------------------------------------------------------------------
constexpr int B_ = 2, N_ = 1024, D_ = 128, H_ = 8, L_ = 8, R_ = 4;
constexpr int HD_ = 16, DH_ = 256;            // head dim, mlp hidden
constexpr long NN_ = (long)N_ * N_;

typedef __attribute__((ext_vector_type(16))) _Float16 f16x16;
typedef __attribute__((ext_vector_type(8)))  float    f32x8;

__device__ inline float gelu_f(float x) {
    return 0.5f * x * (1.0f + erff(x * 0.70710678118654752f));
}
__device__ inline float wave_sum(float v) {
    for (int o = 16; o > 0; o >>= 1) v += __shfl_xor(v, o, 32);
    return v;
}
__device__ inline float wave_max(float v) {
    for (int o = 16; o > 0; o >>= 1) v = fmaxf(v, __shfl_xor(v, o, 32));
    return v;
}

// ---------------------------------------------------------------------------
// Small elementwise kernels
// ---------------------------------------------------------------------------
__global__ void k_f32_to_f16(const float* __restrict__ in, _Float16* __restrict__ out, long n) {
    long i = (long)blockIdx.x * blockDim.x + threadIdx.x;
    if (i < n) out[i] = (_Float16)in[i];
}

__global__ void k_add_masks(const float* __restrict__ a, const float* __restrict__ b,
                            float* __restrict__ o, long n) {
    long i = (long)blockIdx.x * blockDim.x + threadIdx.x;
    if (i < n) o[i] = a[i] + b[i];
}

// rbf[b,i,j,r] = exp(-((dist - mu_r)/sigma_r)^2)
__global__ void k_rbf(const float* __restrict__ dist, const float* __restrict__ mu,
                      const float* __restrict__ sg, float* __restrict__ rbf) {
    long i = (long)blockIdx.x * blockDim.x + threadIdx.x;
    if (i >= (long)B_ * NN_) return;
    float d = dist[i];
    #pragma unroll
    for (int r = 0; r < R_; r++) {
        float t = (d - mu[r]) / sg[r];
        rbf[i * R_ + r] = expf(-t * t);
    }
}

// bias[b,h,i,j] = gelu( LN8([rbf, edge_emb[edge_type]]) @ de_w^T + de_b )
__global__ void k_edge_bias(const float* __restrict__ rbf, const int* __restrict__ et,
                            const float* __restrict__ eemb,  // [4,R] layer slice
                            const float* __restrict__ lng, const float* __restrict__ lnb,
                            const float* __restrict__ dw,    // [H, 2R]
                            const float* __restrict__ db,    // [H]
                            float* __restrict__ bias) {
    long idx = (long)blockIdx.x * blockDim.x + threadIdx.x;
    if (idx >= (long)B_ * NN_) return;
    int  b = (int)(idx / NN_);
    long r = idx % NN_;
    float f[2 * R_];
    const float* rb = rbf + idx * R_;
    #pragma unroll
    for (int c = 0; c < R_; c++) f[c] = rb[c];
    int e = et[idx];
    #pragma unroll
    for (int c = 0; c < R_; c++) f[R_ + c] = eemb[e * R_ + c];
    float m = 0.f;
    #pragma unroll
    for (int c = 0; c < 8; c++) m += f[c];
    m *= 0.125f;
    float v = 0.f;
    #pragma unroll
    for (int c = 0; c < 8; c++) { float d = f[c] - m; v += d * d; }
    float rstd = rsqrtf(v * 0.125f + 1e-5f);
    float nf[8];
    #pragma unroll
    for (int c = 0; c < 8; c++) nf[c] = (f[c] - m) * rstd * lng[c] + lnb[c];
    #pragma unroll
    for (int h = 0; h < H_; h++) {
        float a = db[h];
        #pragma unroll
        for (int c = 0; c < 8; c++) a += dw[h * 8 + c] * nf[c];
        bias[(long)(b * H_ + h) * NN_ + r] = gelu_f(a);
    }
}

// LayerNorm over D=128, fp32 in -> f16 out. One wave per row, 8 rows / block.
__global__ void k_ln128(const float* __restrict__ X, const float* __restrict__ g,
                        const float* __restrict__ bt, _Float16* __restrict__ out, int rows) {
    int lane = threadIdx.x & 31;
    int row  = blockIdx.x * (blockDim.x >> 5) + (threadIdx.x >> 5);
    if (row >= rows) return;
    const float* p = X + (long)row * D_;
    float v[4], s = 0.f;
    #pragma unroll
    for (int t = 0; t < 4; t++) { v[t] = p[lane + 32 * t]; s += v[t]; }
    float mean = wave_sum(s) * (1.f / D_);
    float q = 0.f;
    #pragma unroll
    for (int t = 0; t < 4; t++) { float d = v[t] - mean; q += d * d; }
    float rstd = rsqrtf(wave_sum(q) * (1.f / D_) + 1e-5f);
    #pragma unroll
    for (int t = 0; t < 4; t++) {
        int c = lane + 32 * t;
        out[(long)row * D_ + c] = (_Float16)((v[t] - mean) * rstd * g[c] + bt[c]);
    }
}

// ---------------------------------------------------------------------------
// Fused talking-heads: premix -> softmax -> postmix, all staged through LDS.
// One 256-thread workgroup (8 waves) owns one (b, i) query row:
//   stage 1: wave h async-copies S[b,h,i,0..1023] memory->LDS directly via
//            GLOBAL_LOAD_ASYNC_TO_LDS_B128 (no VGPR round-trip, ASYNCcnt),
//            fenced with s_wait_asynccnt 0 + workgroup barrier   (32 KB LDS)
//   stage 2: wave g premixes sum_h wpre[g,h]*LDS[h][:] into registers,
//            wave32 softmax over 1024 cols, writes back to LDS
//   stage 3: wave g postmixes sum_h wpost[g,h]*LDS[h][:] -> f16 probs
// Replaces three full [B,H,N,N] read+write passes with one read + one f16 write.
// ---------------------------------------------------------------------------
__global__ void k_mix_softmax(const float* __restrict__ S, _Float16* __restrict__ P,
                              const float* __restrict__ wpre, const float* __restrict__ wpost) {
    __shared__ float lds[H_][N_];             // 8 x 1024 f32 = 32 KB
    int lane = threadIdx.x & 31;
    int wv   = threadIdx.x >> 5;              // 0..7, acts as head index
    long bi  = blockIdx.x;                    // b*N + i
    int  b   = (int)(bi >> 10);
    long i   = bi & (N_ - 1);

    const float* src = S + ((long)b * H_ + wv) * NN_ + i * N_;

    // ---- stage 1: async memory -> LDS (CDNA5 GLOBAL_LOAD_ASYNC_TO_LDS_B128)
    // Each lane moves 16 B per issue; 8 issues cover the 4 KB row per wave.
    // LDS byte offset obtained via flat->LDS addrspace cast (low-32 truncation).
    typedef __attribute__((address_space(3))) float lds_f32;
    uint32_t ldst  = (uint32_t)(size_t)(lds_f32*)(&lds[wv][0]) + (uint32_t)lane * 16u;
    uint64_t gsrc  = (uint64_t)(size_t)src + (uint64_t)lane * 16u;
    #pragma unroll
    for (int t = 0; t < 8; t++) {
        asm volatile("global_load_async_to_lds_b128 %0, %1, off"
                     :: "v"(ldst), "v"(gsrc) : "memory");
        ldst += 512u;
        gsrc += 512u;
    }
    asm volatile("s_wait_asynccnt 0x0" ::: "memory");
    __syncthreads();

    // ---- stage 2: pre-softmax head mix + wave32 softmax
    float w[H_];
    #pragma unroll
    for (int h = 0; h < H_; h++) w[h] = wpre[wv * H_ + h];
    float v[32], mx = -3.4e38f;
    #pragma unroll
    for (int t = 0; t < 32; t++) {
        int c = lane + 32 * t;
        float a = 0.f;
        #pragma unroll
        for (int h = 0; h < H_; h++) a += w[h] * lds[h][c];
        v[t] = a;
        mx = fmaxf(mx, a);
    }
    mx = wave_max(mx);
    float s = 0.f;
    #pragma unroll
    for (int t = 0; t < 32; t++) { v[t] = __expf(v[t] - mx); s += v[t]; }
    float inv = 1.f / wave_sum(s);
    __syncthreads();                          // all premix reads done
    #pragma unroll
    for (int t = 0; t < 32; t++) lds[wv][lane + 32 * t] = v[t] * inv;
    __syncthreads();

    // ---- stage 3: post-softmax head mix -> f16 probabilities
    #pragma unroll
    for (int h = 0; h < H_; h++) w[h] = wpost[wv * H_ + h];
    _Float16* dst = P + ((long)b * H_ + wv) * NN_ + i * N_;
    #pragma unroll
    for (int t = 0; t < 32; t++) {
        int c = lane + 32 * t;
        float a = 0.f;
        #pragma unroll
        for (int h = 0; h < H_; h++) a += w[h] * lds[h][c];
        dst[c] = (_Float16)a;
    }
}

// V^T per (b,h): vt[b,h,d,j] = qkv[b,j, 256 + h*16 + d]
__global__ void k_transpose_v(const _Float16* __restrict__ qkv, _Float16* __restrict__ vt) {
    int idx = blockIdx.x * blockDim.x + threadIdx.x;   // B*N*H*HD = 2^18
    int d = idx & 15, h = (idx >> 4) & 7, j = (idx >> 7) & 1023, b = idx >> 17;
    vt[((long)(b * H_ + h) * HD_ + d) * N_ + j] =
        qkv[(long)(b * N_ + j) * (3 * D_) + 2 * D_ + h * HD_ + d];
}

// ---------------------------------------------------------------------------
// Generic WMMA GEMM:  C[m,n] = sum_k A[m,k] * W[n,k]   (fp16 in, fp32 acc)
// One wave computes one 16x16 tile; K stepped by 32 via v_wmma_f32_16x16x32_f16.
// Fragment layouts per CDNA5 ISA 7.12.2:
//   A (16x32 f16): lane m = lane%16; half=lane/16 -> K chunks {half*8..+7, 16+half*8..+7}
//   B (32x16 f16): lane n = lane%16; half -> K chunks {half*8..+7, 16+half*8..+7} (same as A)
//   C (16x16 f32): row = half*8 + r, col = lane%16
// Epilogues: 0 = store f16; 1 = f16 gelu(acc+bias); 2 = X[c] += ls[n]*(acc+bias[n])
// ---------------------------------------------------------------------------
template <int EPI>
__global__ void k_gemm16(const _Float16* __restrict__ A, const _Float16* __restrict__ W,
                         _Float16* __restrict__ C16, float* __restrict__ X,
                         const float* __restrict__ bias, const float* __restrict__ lsv,
                         int M, int K, int lda, int ldw, int ldc,
                         long batchA, long batchW, long cOuter, long cInner, int hcount) {
    int lane = threadIdx.x & 31;
    int half = lane >> 4, ln15 = lane & 15;
    int mt = blockIdx.x * blockDim.y + threadIdx.y;
    if (mt * 16 >= M) return;                 // wave-uniform: EXEC stays full
    int nt = blockIdx.y, z = blockIdx.z;
    const _Float16* Ab = A + (long)z * batchA;
    const _Float16* Wb = W + (long)z * batchW;
    long cb = (long)(z / hcount) * cOuter + (long)(z % hcount) * cInner;
    int arow = mt * 16 + ln15;
    int wrow = nt * 16 + ln15;
    f32x8 acc = {};
    for (int k0 = 0; k0 < K; k0 += 32) {
        union { f16x16 v; uint4 q[2]; } af, bf;
        const uint4* ap = (const uint4*)(Ab + (long)arow * lda + k0);
        af.q[0] = ap[half];
        af.q[1] = ap[2 + half];
        const uint4* wp = (const uint4*)(Wb + (long)wrow * ldw + k0);
        bf.q[0] = wp[half];
        bf.q[1] = wp[2 + half];
        acc = __builtin_amdgcn_wmma_f32_16x16x32_f16(false, af.v, false, bf.v,
                                                     (short)0, acc, false, false);
    }
    #pragma unroll
    for (int r = 0; r < 8; r++) {
        int row = mt * 16 + half * 8 + r;
        int col = nt * 16 + ln15;
        float v = acc[r];
        long ci = cb + (long)row * ldc + col;
        if (EPI == 0)      C16[ci] = (_Float16)v;
        else if (EPI == 1) C16[ci] = (_Float16)gelu_f(v + bias[col]);
        else               X[ci] += lsv[col] * (v + bias[col]);
    }
}

// Scores: S[b,h,i,j] = 0.25 * <q_i, k_j> + mask[b,i,j] + bias[b,h,i,j]
// K = HD = 16, zero-padded to 32 in both fragments.
__global__ void k_scores(const _Float16* __restrict__ qkv, const float* __restrict__ mask,
                         const float* __restrict__ biasT, float* __restrict__ S) {
    int lane = threadIdx.x & 31;
    int half = lane >> 4, ln15 = lane & 15;
    int mt = blockIdx.x * blockDim.y + threadIdx.y;   // i-tile (0..63)
    int nt = blockIdx.y;                              // j-tile (0..63)
    int z  = blockIdx.z;                              // b*H + h
    int b = z >> 3, h = z & 7;
    int i0 = mt * 16, j0 = nt * 16;
    union { f16x16 v; uint4 q[2]; } af, bf;
    const uint4 z4 = make_uint4(0u, 0u, 0u, 0u);
    const uint4* qp = (const uint4*)(qkv + (long)(b * N_ + i0 + ln15) * (3 * D_) + h * HD_);
    af.q[0] = qp[half];          // K 0..7 (half 0) / 8..15 (half 1)
    af.q[1] = z4;                // K 16..31 padded
    const uint4* kp = (const uint4*)(qkv + (long)(b * N_ + j0 + ln15) * (3 * D_) + D_ + h * HD_);
    bf.q[0] = kp[half];          // K 0..7 (half 0) / 8..15 (half 1)
    bf.q[1] = z4;                // K 16..31 padded
    f32x8 acc = {};
    acc = __builtin_amdgcn_wmma_f32_16x16x32_f16(false, af.v, false, bf.v,
                                                 (short)0, acc, false, false);
    #pragma unroll
    for (int r = 0; r < 8; r++) {
        int row = i0 + half * 8 + r;
        int col = j0 + ln15;
        long mi = (long)(b * N_ + row) * N_ + col;
        long bi = (long)(b * H_ + h) * NN_ + (long)row * N_ + col;
        S[bi] = acc[r] * 0.25f + mask[mi] + biasT[bi];
    }
}

// ---------------------------------------------------------------------------
// Host orchestration
// ---------------------------------------------------------------------------
static inline size_t alignup(size_t v) { return (v + 255) & ~(size_t)255; }

extern "C" void kernel_launch(void* const* d_in, const int* in_sizes, int n_in,
                              void* d_out, int out_size, void* d_ws, size_t ws_size,
                              hipStream_t stream) {
    (void)in_sizes; (void)n_in; (void)out_size; (void)ws_size;

    const float* x_in      = (const float*)d_in[0];
    const float* dist      = (const float*)d_in[1];
    const float* attn_mask = (const float*)d_in[2];
    const float* local_mask= (const float*)d_in[3];
    const int*   edge_type = (const int*)  d_in[4];
    const float* ln_g      = (const float*)d_in[5];
    const float* ln_b      = (const float*)d_in[6];
    const float* qkv_w     = (const float*)d_in[7];
    const float* proj_w    = (const float*)d_in[8];
    const float* proj_b    = (const float*)d_in[9];
    const float* th_pre    = (const float*)d_in[10];
    const float* th_post   = (const float*)d_in[11];
    const float* ls        = (const float*)d_in[12];
    const float* fc1_w     = (const float*)d_in[13];
    const float* fc1_b     = (const float*)d_in[14];
    const float* fc2_w     = (const float*)d_in[15];
    const float* fc2_b     = (const float*)d_in[16];
    const float* de_ln_g   = (const float*)d_in[17];
    const float* de_ln_b   = (const float*)d_in[18];
    const float* de_w      = (const float*)d_in[19];
    const float* de_b      = (const float*)d_in[20];
    const float* edge_emb  = (const float*)d_in[21];
    const float* rbf_mu    = (const float*)d_in[22];
    const float* rbf_sigma = (const float*)d_in[23];

    // Workspace carve-up
    char* ws = (char*)d_ws;
    size_t off = 0;
    auto take = [&](size_t bytes) { size_t o = off; off = alignup(off + bytes); return o; };
    float*     rbf   = (float*)    (ws + take((size_t)B_ * NN_ * R_ * 4));
    float*     mask1 = (float*)    (ws + take((size_t)B_ * NN_ * 4));
    float*     biasB = (float*)    (ws + take((size_t)B_ * H_ * NN_ * 4));
    float*     Sb    = (float*)    (ws + take((size_t)B_ * H_ * NN_ * 4));
    _Float16*  P16   = (_Float16*) (ws + take((size_t)B_ * H_ * NN_ * 2));
    _Float16*  xn    = (_Float16*) (ws + take((size_t)B_ * N_ * D_ * 2));
    _Float16*  qkv16 = (_Float16*) (ws + take((size_t)B_ * N_ * 3 * D_ * 2));
    _Float16*  vt    = (_Float16*) (ws + take((size_t)B_ * H_ * HD_ * N_ * 2));
    _Float16*  o16   = (_Float16*) (ws + take((size_t)B_ * N_ * D_ * 2));
    _Float16*  h1    = (_Float16*) (ws + take((size_t)B_ * N_ * DH_ * 2));
    _Float16*  wq16  = (_Float16*) (ws + take((size_t)L_ * 2 * 3 * D_ * D_ * 2));
    _Float16*  wp16  = (_Float16*) (ws + take((size_t)L_ * 2 * D_ * D_ * 2));
    _Float16*  w116  = (_Float16*) (ws + take((size_t)L_ * DH_ * D_ * 2));
    _Float16*  w216  = (_Float16*) (ws + take((size_t)L_ * D_ * DH_ * 2));

    float* xout = (float*)d_out;   // running residual stream, fp32

    // x -> out (residual stream lives in d_out)
    hipMemcpyAsync(xout, x_in, (size_t)B_ * N_ * D_ * sizeof(float),
                   hipMemcpyDeviceToDevice, stream);

    // Weight precision conversion (fp32 -> fp16)
    {
        long n;
        n = (long)L_ * 2 * 3 * D_ * D_;
        k_f32_to_f16<<<(n + 255) / 256, 256, 0, stream>>>(qkv_w, wq16, n);
        n = (long)L_ * 2 * D_ * D_;
        k_f32_to_f16<<<(n + 255) / 256, 256, 0, stream>>>(proj_w, wp16, n);
        n = (long)L_ * DH_ * D_;
        k_f32_to_f16<<<(n + 255) / 256, 256, 0, stream>>>(fc1_w, w116, n);
        n = (long)L_ * D_ * DH_;
        k_f32_to_f16<<<(n + 255) / 256, 256, 0, stream>>>(fc2_w, w216, n);
    }
    // first_mask = attn_mask + local_mask ; RBF features
    k_add_masks<<<(B_ * NN_ + 255) / 256, 256, 0, stream>>>(attn_mask, local_mask, mask1, B_ * NN_);
    k_rbf<<<(B_ * NN_ + 255) / 256, 256, 0, stream>>>(dist, rbf_mu, rbf_sigma, rbf);

    const int ROWS = B_ * N_;   // 2048
    const dim3 wmmaBlk(32, 4);

    for (int l = 0; l < L_; l++) {
        // per-layer pairwise bias [B,H,N,N]
        k_edge_bias<<<(B_ * NN_ + 255) / 256, 256, 0, stream>>>(
            rbf, edge_type, edge_emb + l * 4 * R_, de_ln_g + l * 8, de_ln_b + l * 8,
            de_w + l * H_ * 8, de_b + l * H_, biasB);

        for (int s = 0; s < 2; s++) {
            const float* mask = (s == 0) ? mask1 : attn_mask;
            // LN -> f16
            k_ln128<<<ROWS / 8, 256, 0, stream>>>(
                xout, ln_g + (l * 3 + s) * D_, ln_b + (l * 3 + s) * D_, xn, ROWS);
            // qkv = xn @ qkv_w^T   [2048 x 384]
            k_gemm16<0><<<dim3(32, 24, 1), wmmaBlk, 0, stream>>>(
                xn, wq16 + (long)(l * 2 + s) * 3 * D_ * D_, qkv16, nullptr, nullptr, nullptr,
                ROWS, D_, D_, D_, 3 * D_, 0, 0, 0, 0, 1);
            k_transpose_v<<<(B_ * N_ * H_ * HD_) / 256, 256, 0, stream>>>(qkv16, vt);
            // S = 0.25*QK^T + mask + bias   [B,H,N,N]
            k_scores<<<dim3(16, 64, B_ * H_), wmmaBlk, 0, stream>>>(qkv16, mask, biasB, Sb);
            // fused: premix -> softmax -> postmix (probs straight to f16)
            k_mix_softmax<<<B_ * N_, 256, 0, stream>>>(
                Sb, P16, th_pre + (l * 2 + s) * H_ * H_, th_post + (l * 2 + s) * H_ * H_);
            // o = P @ V  (per b,h batch; scatter heads into [B,N,128])
            k_gemm16<0><<<dim3(16, 1, B_ * H_), wmmaBlk, 0, stream>>>(
                P16, vt, o16, nullptr, nullptr, nullptr,
                N_, N_, N_, N_, D_,
                NN_, (long)HD_ * N_, (long)N_ * D_, HD_, H_);
            // x += ls * (o @ proj_w^T + proj_b)
            k_gemm16<2><<<dim3(32, 8, 1), wmmaBlk, 0, stream>>>(
                o16, wp16 + (long)(l * 2 + s) * D_ * D_, nullptr, xout,
                proj_b + (l * 2 + s) * D_, ls + (l * 3 + s) * D_,
                ROWS, D_, D_, D_, D_, 0, 0, 0, 0, 1);
        }
        // MLP: x += ls2 * (gelu(LN(x)@fc1^T + b1) @ fc2^T + b2)
        k_ln128<<<ROWS / 8, 256, 0, stream>>>(
            xout, ln_g + (l * 3 + 2) * D_, ln_b + (l * 3 + 2) * D_, xn, ROWS);
        k_gemm16<1><<<dim3(32, 16, 1), wmmaBlk, 0, stream>>>(
            xn, w116 + (long)l * DH_ * D_, h1, nullptr, fc1_b + l * DH_, nullptr,
            ROWS, D_, D_, D_, DH_, 0, 0, 0, 0, 1);
        k_gemm16<2><<<dim3(32, 8, 1), wmmaBlk, 0, stream>>>(
            h1, w216 + (long)l * D_ * DH_, nullptr, xout, fc2_b + l * D_, ls + (l * 3 + 2) * D_,
            ROWS, DH_, DH_, DH_, D_, 0, 0, 0, 0, 1);
    }
}
